// HGNN_AC_40570261078600
// MI455X (gfx1250) — matrix operand order repaired and checked by reading, not compiled, not run on hardware
//
#include <hip/hip_runtime.h>
#include <hip/hip_bf16.h>
#include <cstdint>

typedef __attribute__((ext_vector_type(16))) _Float16 v16h;
typedef __attribute__((ext_vector_type(8)))  _Float16 v8h;
typedef __attribute__((ext_vector_type(8)))  float    v8f;

#define HID   64
#define NHEAD 8
#define PADR  72   // padded LDS row (halves): 144B stride -> conflict-free b128 reads

// D = A(16x32 f16) * B(32x16 f16) + C(16x16 f32)
__device__ __forceinline__ v8f wmma16(v16h a, v16h b, v8f c) {
  return __builtin_amdgcn_wmma_f32_16x16x32_f16(false, a, false, b, (short)0, c,
                                                false, false);
}

// A-fragment: lane holds row M=lane%16, halves [kbase..kbase+15] (contiguous)
__device__ __forceinline__ v16h load_afrag(const _Float16* p) {
  v8h lo = *(const v8h*)p;
  v8h hi = *(const v8h*)(p + 8);
  return __builtin_shufflevector(lo, hi, 0, 1, 2, 3, 4, 5, 6, 7,
                                         8, 9, 10, 11, 12, 13, 14, 15);
}
__device__ __forceinline__ v16h load_afrag(const float* p) {  // with f32->f16 convert
  v16h r;
#pragma unroll
  for (int i = 0; i < 8; ++i) {
    r[i]     = (_Float16)p[i];
    r[i + 8] = (_Float16)p[i + 8];
  }
  return r;
}

// ---- CDNA5 async copy: global -> LDS, 16B per lane, tracked by ASYNCcnt ----
__device__ __forceinline__ uint32_t lds_off(const void* p) {
  return (uint32_t)(uintptr_t)p;   // low 32 bits of generic LDS ptr = LDS address
}
__device__ __forceinline__ void async_ld_b128(uint32_t dst_lds, const void* gsrc) {
  asm volatile("global_load_async_to_lds_b128 %0, %1, off"
               :: "v"(dst_lds), "v"((uint64_t)(uintptr_t)gsrc)
               : "memory");
}
__device__ __forceinline__ void wait_async0() {
  asm volatile("s_wait_asynccnt 0x0" ::: "memory");
}

// ---------------------------------------------------------------------------
// Weight / V prep: f32 -> f16 with transpose so B-fragments load contiguously.
// ---------------------------------------------------------------------------
__global__ void prep_weights(const float* __restrict__ W, const float* __restrict__ W2,
                             _Float16* __restrict__ Wt, _Float16* __restrict__ W2t) {
  int i = blockIdx.x * blockDim.x + threadIdx.x;
  if (i >= NHEAD * HID * HID) return;
  int h = i >> 12, r = (i >> 6) & 63, c = i & 63;
  int src = (h << 12) + (c << 6) + r;   // transposed element
  Wt[i]  = (_Float16)W[src];
  W2t[i] = (_Float16)W2[src];
}

__global__ void prep_v(const float* __restrict__ F, _Float16* __restrict__ Vt, int NS) {
  int i = blockIdx.x * blockDim.x + threadIdx.x;
  if (i >= NS * HID) return;
  int f = i / NS, s = i - f * NS;
  Vt[i] = (_Float16)F[s * HID + f];
}

// ---------------------------------------------------------------------------
// Projection GEMM: out[h] = A[N x 64] @ Bt[h]^T (Bt is [h][64 out][64 k], k-contig)
// ---------------------------------------------------------------------------
template <typename AT>
__global__ __launch_bounds__(128, 1)
void proj_kernel(const AT* __restrict__ A, const _Float16* __restrict__ Bt,
                 _Float16* __restrict__ out, int N) {
  const int h    = blockIdx.y;
  const int wave = threadIdx.x >> 5;
  const int lane = threadIdx.x & 31;
  const int half = lane >> 4;
  const int l16  = lane & 15;
  const int row0 = blockIdx.x * 64 + wave * 16;

  v16h af[2];
#pragma unroll
  for (int kc = 0; kc < 2; ++kc)
    af[kc] = load_afrag(A + (size_t)(row0 + l16) * HID + kc * 32 + half * 16);

  const _Float16* Bh = Bt + (size_t)h * HID * HID;
  v8f acc[4];
#pragma unroll
  for (int nt = 0; nt < 4; ++nt) {
    v8f z = {};
#pragma unroll
    for (int kc = 0; kc < 2; ++kc) {
      v16h bf = *(const v16h*)(Bh + (size_t)(nt * 16 + l16) * HID + kc * 32 + half * 16);
      z = wmma16(af[kc], bf, z);
    }
    acc[nt] = z;
  }

  _Float16* Oh = out + (size_t)h * N * HID;
#pragma unroll
  for (int nt = 0; nt < 4; ++nt)
#pragma unroll
    for (int v = 0; v < 8; ++v)
      Oh[(size_t)(row0 + v + 8 * half) * HID + nt * 16 + l16] = (_Float16)acc[nt][v];
}

// ---------------------------------------------------------------------------
// Flash attention per (head, 64 dest rows). Each wave owns 16 dest rows.
// K/V tiles staged once per block via async global->LDS copies (double-buffered).
// ---------------------------------------------------------------------------
__global__ __launch_bounds__(128, 1)
void attn_kernel(const _Float16* __restrict__ Qp, const _Float16* __restrict__ Kb,
                 const _Float16* __restrict__ Vt, float* __restrict__ Hout,
                 int ND, int NS) {
  __shared__ __attribute__((aligned(16))) _Float16 sK[2][64 * PADR];
  __shared__ __attribute__((aligned(16))) _Float16 sV[2][64 * PADR];
  __shared__ __attribute__((aligned(16))) _Float16 sP[4][16 * PADR];

  const int h    = blockIdx.y;
  const int tid  = threadIdx.x;
  const int wave = tid >> 5;
  const int lane = tid & 31;
  const int half = lane >> 4;
  const int l16  = lane & 15;
  const int n0   = blockIdx.x * 64 + wave * 16;
  _Float16* smP  = &sP[wave][0];

  const _Float16* Qh     = Qp + (size_t)h * ND * HID;
  const _Float16* KhBase = Kb + (size_t)h * NS * HID;   // rows contiguous (stride HID)

  // cooperative async tile copy: 1024 x b128 per tile pair, 8 per thread
  auto issue_tile = [&](int s0, int buf) {
    const _Float16* kg = KhBase + (size_t)s0 * HID;     // 64x64 contiguous block
#pragma unroll
    for (int c = 0; c < 4; ++c) {
      int i   = tid + 128 * c;       // 0..511
      int row = i >> 3, ch = (i & 7) * 8;
      async_ld_b128(lds_off(&sK[buf][row * PADR + ch]), kg + row * HID + ch);
      async_ld_b128(lds_off(&sV[buf][row * PADR + ch]), Vt + (size_t)row * NS + s0 + ch);
    }
  };

  v16h qf[2];
#pragma unroll
  for (int kc = 0; kc < 2; ++kc)
    qf[kc] = load_afrag(Qh + (size_t)(n0 + l16) * HID + kc * 32 + half * 16);

  v8f acc_o[4];
  v8f zf = {};
#pragma unroll
  for (int ft = 0; ft < 4; ++ft) acc_o[ft] = zf;
  float m_i[8], l_i[8];
#pragma unroll
  for (int v = 0; v < 8; ++v) { m_i[v] = -1e30f; l_i[v] = 0.0f; }

  const int nIter = NS / 64;
  issue_tile(0, 0);

#pragma unroll 1
  for (int it = 0; it < nIter; ++it) {
    wait_async0();        // my slice of tile `it` has landed in LDS
    __syncthreads();      // everyone's slice landed; prev buffer fully consumed
    if (it + 1 < nIter) issue_tile((it + 1) * 64, (it + 1) & 1);

    const _Float16* Kl = &sK[it & 1][0];
    const _Float16* Vl = &sV[it & 1][0];

    // ---- scores e[16 dest x 64 src] : 8 WMMAs, B-frags from LDS ----
    v8f e[4];
#pragma unroll
    for (int st = 0; st < 4; ++st) {
      const _Float16* kp = Kl + (st * 16 + l16) * PADR + half * 16;
      v16h b0 = *(const v16h*)kp;
      v16h b1 = *(const v16h*)(kp + 32);
      v8f z = {};
      z = wmma16(qf[0], b0, z);
      e[st] = wmma16(qf[1], b1, z);
    }

    // ---- leakyReLU (= max(x, 0.2x)) + online softmax; scatter P (f16) to LDS ----
#pragma unroll
    for (int v = 0; v < 8; ++v) {
      float mx = -1e30f;
#pragma unroll
      for (int st = 0; st < 4; ++st) {
        float x = e[st][v];
        x = fmaxf(x, 0.2f * x);
        e[st][v] = x;
        mx = fmaxf(mx, x);
      }
#pragma unroll
      for (int msk = 1; msk < 16; msk <<= 1)
        mx = fmaxf(mx, __shfl_xor(mx, msk, 16));
      float mN = fmaxf(m_i[v], mx);
      float sc = __expf(m_i[v] - mN);
      m_i[v] = mN;
      float rs = 0.0f;
#pragma unroll
      for (int st = 0; st < 4; ++st) {
        float p = __expf(e[st][v] - mN);
        rs += p;
        smP[(v + 8 * half) * PADR + st * 16 + l16] = (_Float16)p;
      }
#pragma unroll
      for (int msk = 1; msk < 16; msk <<= 1)
        rs += __shfl_xor(rs, msk, 16);
      l_i[v] = l_i[v] * sc + rs;
#pragma unroll
      for (int ft = 0; ft < 4; ++ft) acc_o[ft][v] *= sc;
    }

    // cross-lane LDS RAW within the wave: wait for DS stores before re-reading
    asm volatile("s_wait_dscnt 0x0" ::: "memory");

    v16h pf[2];
#pragma unroll
    for (int kc = 0; kc < 2; ++kc)
      pf[kc] = load_afrag(smP + l16 * PADR + kc * 32 + half * 16);

    // ---- O += P @ V : 8 WMMAs, B-frags from LDS ----
#pragma unroll
    for (int ft = 0; ft < 4; ++ft) {
      const _Float16* vp = Vl + (ft * 16 + l16) * PADR + half * 16;
      v16h v0 = *(const v16h*)vp;
      v16h v1 = *(const v16h*)(vp + 32);
      acc_o[ft] = wmma16(pf[0], v0, acc_o[ft]);
      acc_o[ft] = wmma16(pf[1], v1, acc_o[ft]);
    }
  }

  // ---- normalize, ELU, per-head store ----
  float* Oh = Hout + (size_t)h * ND * HID;
#pragma unroll
  for (int v = 0; v < 8; ++v) {
    float inv = 1.0f / l_i[v];
#pragma unroll
    for (int ft = 0; ft < 4; ++ft) {
      float x = acc_o[ft][v] * inv;
      x = (x > 0.0f) ? x : (__expf(x) - 1.0f);
      Oh[(size_t)(n0 + v + 8 * half) * HID + ft * 16 + l16] = x;
    }
  }
}

// out = mean over heads (deterministic reduce, no float atomics)
__global__ void reduce_mean(const float* __restrict__ Hb, float* __restrict__ out, int ND) {
  int i = blockIdx.x * blockDim.x + threadIdx.x;
  if (i >= ND * HID) return;
  float s = 0.0f;
#pragma unroll
  for (int h = 0; h < NHEAD; ++h) s += Hb[(size_t)h * ND * HID + i];
  out[i] = s * (1.0f / NHEAD);
}

extern "C" void kernel_launch(void* const* d_in, const int* in_sizes, int n_in,
                              void* d_out, int out_size, void* d_ws, size_t ws_size,
                              hipStream_t stream) {
  (void)n_in; (void)out_size; (void)ws_size;
  const float* emb_dest = (const float*)d_in[0];
  const float* emb_src  = (const float*)d_in[1];
  const float* feat_src = (const float*)d_in[2];
  const float* W        = (const float*)d_in[3];
  const float* W2       = (const float*)d_in[4];
  float* out = (float*)d_out;
  const int ND = in_sizes[0] / HID;
  const int NS = in_sizes[1] / HID;

  char* ws = (char*)d_ws;
  size_t off = 0;
  auto alloc = [&](size_t bytes) -> char* {
    char* p = ws + off;
    off += (bytes + 255) & ~(size_t)255;
    return p;
  };
  _Float16* Kbuf = (_Float16*)alloc((size_t)NHEAD * NS * HID * 2);  // h1 per head
  _Float16* Qp   = (_Float16*)alloc((size_t)NHEAD * ND * HID * 2);  // t per head
  _Float16* h2b  = (_Float16*)alloc((size_t)NHEAD * ND * HID * 2);  // h2 per head
  _Float16* Vt   = (_Float16*)alloc((size_t)HID * NS * 2);          // feat^T
  _Float16* Wt   = (_Float16*)alloc((size_t)NHEAD * HID * HID * 2);
  _Float16* W2t  = (_Float16*)alloc((size_t)NHEAD * HID * HID * 2);
  float*    Hb   = (float*)alloc((size_t)NHEAD * ND * HID * 4);     // elu(h') per head

  prep_weights<<<(NHEAD * HID * HID + 255) / 256, 256, 0, stream>>>(W, W2, Wt, W2t);
  prep_v<<<(NS * HID + 255) / 256, 256, 0, stream>>>(feat_src, Vt, NS);

  proj_kernel<float><<<dim3(NS / 64, NHEAD), 128, 0, stream>>>(emb_src, Wt, Kbuf, NS);
  proj_kernel<float><<<dim3(ND / 64, NHEAD), 128, 0, stream>>>(emb_dest, Wt, h2b, ND);
  proj_kernel<_Float16><<<dim3(ND / 64, NHEAD), 128, 0, stream>>>(h2b, W2t, Qp, ND);

  attn_kernel<<<dim3(ND / 64, NHEAD), 128, 0, stream>>>(Qp, Kbuf, Vt, Hb, ND, NS);
  reduce_mean<<<(ND * HID + 255) / 256, 256, 0, stream>>>(Hb, out, ND);
}
